// HANClass_31825707664042
// MI455X (gfx1250) — matrix operand order repaired and checked by reading, not compiled, not run on hardware
//
#include <hip/hip_runtime.h>
#include <hip/hip_bf16.h>

// ---------------------------------------------------------------------------
// HAN (heterogeneous graph attention) forward for MI455X / gfx1250.
//   - Dense GEMMs on v_wmma_f32_16x16x32_f16: block tile 128x64, wave tile
//     32x32 (4 WMMA / K-step), double-buffered LDS, fully vectorized staging
//     (global_load_b128 -> v_cvt_pk_f16_f32 -> ds_store, ds_load_b128 frags),
//     B weights pre-transposed to [N,K] f16 so B-fragments are contiguous.
//   - Row index clamped in staging (epilogues guard M) -> no exec-mask
//     branches in the hot loop.
//   - Semantic attention tanh(stk@kW+kb).mean(0) fused into the WMMA epilogue.
//   - GAT softmax over in-edges via float atomic max (CAS) + atomic add.
// ---------------------------------------------------------------------------

typedef __attribute__((ext_vector_type(16))) _Float16 v16h;
typedef __attribute__((ext_vector_type(8)))  _Float16 v8h;
typedef __attribute__((ext_vector_type(4)))  _Float16 v4h;
typedef __attribute__((ext_vector_type(8)))  float    v8f;

#define N_ADDR 100000
#define N_TX   200000
#define F_IN   128
#define HID    256
#define OUT_C  128
#define HEADS  8
#define N_EDGE 250000
#define NCLS   2

// ------------------------------- utility kernels ---------------------------

__global__ void fill_kernel(float* __restrict__ p, float v, size_t n) {
  size_t i = (size_t)blockIdx.x * blockDim.x + threadIdx.x;
  if (i < n) p[i] = v;
}

// d[N,K] = (f16) s[K,N]  (weight transpose, one-time tiny kernel)
__global__ void f32_to_f16_T_kernel(const float* __restrict__ s,
                                    _Float16* __restrict__ d, int K, int N) {
  int j = blockIdx.x * blockDim.x + threadIdx.x;
  if (j >= N * K) return;
  int n = j / K, k = j % K;
  d[j] = (_Float16)s[(size_t)k * N + n];
}

__global__ void relu_inplace_kernel(float* __restrict__ p, size_t n) {
  size_t i = (size_t)blockIdx.x * blockDim.x + threadIdx.x;
  if (i < n) { float v = p[i]; p[i] = v > 0.f ? v : 0.f; }
}

__global__ void fixmax_kernel(float* __restrict__ p, size_t n) {
  size_t i = (size_t)blockIdx.x * blockDim.x + threadIdx.x;
  if (i < n) { float v = p[i]; if (!(v > -1e37f)) p[i] = 0.f; }
}

__device__ __forceinline__ void atomicMaxF(float* addr, float val) {
  unsigned int* ua = (unsigned int*)addr;
  unsigned int old = *ua;
  while (__uint_as_float(old) < val) {
    unsigned int assumed = old;
    old = atomicCAS(ua, assumed, __float_as_uint(val));
    if (old == assumed) break;
  }
}

// --------------------------- WMMA GEMM core --------------------------------
// Block = 256 threads = 8 waves (wave32). Block tile 128x64, wave tile 32x32.
// A fp32 row-major [M,K]; BT f16 [N,K] (transposed); K % 32 == 0, N % 64 == 0.
// Double-buffered LDS; row index clamped (epilogues guard gm < M).

#define GEMM_DECLS()                                                           \
  __shared__ _Float16 As[2][128][40];                                          \
  __shared__ _Float16 Bs[2][64][40];                                           \
  const int t = threadIdx.x;                                                   \
  const int lane = t & 31, wave = t >> 5;                                      \
  const int row0 = blockIdx.x * 128, col0 = blockIdx.y * 64;                   \
  const int wr = (wave & 3) * 32;                                              \
  const int wc = (wave >> 2) * 32;                                             \
  const int hi = lane >> 4, ml = lane & 15;                                    \
  const int hi8 = hi << 3, hi16 = hi << 4;                                     \
  const int sar = t >> 3, sac = (t & 7) << 2;    /* A staging row/col */       \
  const int sbc = t >> 2, sbk = (t & 3) << 3;    /* B staging col/k   */       \
  v8f acc00 = {}, acc01 = {}, acc10 = {}, acc11 = {};                          \
  float4 fa[4];                                                                \
  v8h fb;

#define STAGE_GLOBAL(K0)                                                       \
  _Pragma("unroll")                                                            \
  for (int u = 0; u < 4; ++u) {                                                \
    int gr = row0 + sar + u * 32;                                              \
    gr = gr < M ? gr : M - 1;                 /* clamp; epilogue guards */     \
    fa[u] = *(const float4*)(A + (size_t)gr * K + (K0) + sac);                 \
  }                                                                            \
  fb = *(const v8h*)(BT + (size_t)(col0 + sbc) * K + (K0) + sbk);

#define STAGE_STORE(BUF)                                                       \
  _Pragma("unroll")                                                            \
  for (int u = 0; u < 4; ++u) {                                                \
    v4h hq;                                                                    \
    hq[0] = (_Float16)fa[u].x; hq[1] = (_Float16)fa[u].y;                      \
    hq[2] = (_Float16)fa[u].z; hq[3] = (_Float16)fa[u].w;                      \
    *(v4h*)&As[BUF][sar + u * 32][sac] = hq;                                   \
  }                                                                            \
  *(v8h*)&Bs[BUF][sbc][sbk] = fb;

#define FRAGS_WMMA(BUF)                                                        \
  {                                                                            \
    v8h a0lo = *(const v8h*)&As[BUF][wr + ml][hi8];                            \
    v8h a0hi = *(const v8h*)&As[BUF][wr + ml][16 + hi8];                       \
    v8h a1lo = *(const v8h*)&As[BUF][wr + 16 + ml][hi8];                       \
    v8h a1hi = *(const v8h*)&As[BUF][wr + 16 + ml][16 + hi8];                  \
    v8h b0lo = *(const v8h*)&Bs[BUF][wc + ml][hi8];                            \
    v8h b0hi = *(const v8h*)&Bs[BUF][wc + ml][16 + hi8];                       \
    v8h b1lo = *(const v8h*)&Bs[BUF][wc + 16 + ml][hi8];                       \
    v8h b1hi = *(const v8h*)&Bs[BUF][wc + 16 + ml][16 + hi8];                  \
    v16h a0 = __builtin_shufflevector(a0lo, a0hi, 0,1,2,3,4,5,6,7,8,9,10,11,12,13,14,15); \
    v16h a1 = __builtin_shufflevector(a1lo, a1hi, 0,1,2,3,4,5,6,7,8,9,10,11,12,13,14,15); \
    v16h b0 = __builtin_shufflevector(b0lo, b0hi, 0,1,2,3,4,5,6,7,8,9,10,11,12,13,14,15); \
    v16h b1 = __builtin_shufflevector(b1lo, b1hi, 0,1,2,3,4,5,6,7,8,9,10,11,12,13,14,15); \
    acc00 = __builtin_amdgcn_wmma_f32_16x16x32_f16(false, a0, false, b0, (short)0, acc00, false, false); \
    acc01 = __builtin_amdgcn_wmma_f32_16x16x32_f16(false, a0, false, b1, (short)0, acc01, false, false); \
    acc10 = __builtin_amdgcn_wmma_f32_16x16x32_f16(false, a1, false, b0, (short)0, acc10, false, false); \
    acc11 = __builtin_amdgcn_wmma_f32_16x16x32_f16(false, a1, false, b1, (short)0, acc11, false, false); \
  }

// Prologue fills buffer 0; each iteration prefetches chunk k+1 (global loads
// issue before the WMMAs, s_wait_loadcnt lands after them), one barrier/iter.
#define GEMM_MAINLOOP()                                                        \
  STAGE_GLOBAL(0);                                                             \
  STAGE_STORE(0);                                                              \
  __syncthreads();                                                             \
  {                                                                            \
    int cur = 0;                                                               \
    for (int k0 = 0; k0 < K; k0 += 32) {                                       \
      const int kn = k0 + 32;                                                  \
      const bool hn = kn < K;                                                  \
      if (hn) { STAGE_GLOBAL(kn); }                                            \
      FRAGS_WMMA(cur);                                                         \
      if (hn) { STAGE_STORE(cur ^ 1); }                                        \
      __syncthreads();                                                         \
      cur ^= 1;                                                                \
    }                                                                          \
  }

// C[M,N] = act(A @ BT^T + bias)
__global__ void __launch_bounds__(256)
gemm_wmma_kernel(const float* __restrict__ A, const _Float16* __restrict__ BT,
                 const float* __restrict__ bias, float* __restrict__ C,
                 int M, int N, int K, int relu) {
  GEMM_DECLS();
  GEMM_MAINLOOP();

  int c0 = col0 + wc + ml, c1 = c0 + 16;
  float bz0 = bias[c0], bz1 = bias[c1];
#pragma unroll
  for (int r = 0; r < 8; ++r) {              // ISA C/D layout: m = r + 8*hi
    int gm0 = row0 + wr + hi8 + r;
    int gm1 = gm0 + 16;
    if (gm0 < M) {
      float v0 = acc00[r] + bz0, v1 = acc01[r] + bz1;
      if (relu) { v0 = v0 > 0.f ? v0 : 0.f; v1 = v1 > 0.f ? v1 : 0.f; }
      C[(size_t)gm0 * N + c0] = v0;
      C[(size_t)gm0 * N + c1] = v1;
    }
    if (gm1 < M) {
      float v0 = acc10[r] + bz0, v1 = acc11[r] + bz1;
      if (relu) { v0 = v0 > 0.f ? v0 : 0.f; v1 = v1 > 0.f ? v1 : 0.f; }
      C[(size_t)gm1 * N + c0] = v0;
      C[(size_t)gm1 * N + c1] = v1;
    }
  }
}

// colsum[c] += sum_n tanh( (A@kW)[n,c] + kb[c] )   (tanh matrix never stored)
__global__ void __launch_bounds__(256)
gemm_tanh_colsum_kernel(const float* __restrict__ A, const _Float16* __restrict__ BT,
                        const float* __restrict__ kbias, float* __restrict__ colsum,
                        int M, int N, int K) {
  __shared__ float colacc[64];
  GEMM_DECLS();
  if (t < 64) colacc[t] = 0.f;
  GEMM_MAINLOOP();

  int c0 = col0 + wc + ml, c1 = c0 + 16;
  float bz0 = kbias[c0], bz1 = kbias[c1];
  float p0 = 0.f, p1 = 0.f;
#pragma unroll
  for (int r = 0; r < 8; ++r) {
    int gm0 = row0 + wr + hi8 + r;
    int gm1 = gm0 + 16;
    if (gm0 < M) { p0 += tanhf(acc00[r] + bz0); p1 += tanhf(acc01[r] + bz1); }
    if (gm1 < M) { p0 += tanhf(acc10[r] + bz0); p1 += tanhf(acc11[r] + bz1); }
  }
  atomicAdd(&colacc[wc + ml], p0);
  atomicAdd(&colacc[wc + 16 + ml], p1);
  __syncthreads();
  if (t < 64) atomicAdd(&colsum[col0 + t], colacc[t]);
}

// --------------------------- GAT message passing ---------------------------

// out[n,h] = sum_d h[n, h*D+d] * att[h*D+d]
__global__ void node_att_kernel(const float* __restrict__ h,
                                const float* __restrict__ att,
                                float* __restrict__ out, int N, int D) {
  int idx = blockIdx.x * blockDim.x + threadIdx.x;
  if (idx >= N * HEADS) return;
  int n = idx / HEADS, hh = idx % HEADS;
  const float* hp = h + (size_t)n * HEADS * D + hh * D;
  const float* ap = att + hh * D;
  float s = 0.f;
  for (int d = 0; d < D; ++d) s += hp[d] * ap[d];
  out[idx] = s;
}

__global__ void edge_alpha_max_kernel(const int* __restrict__ src,
                                      const int* __restrict__ dst,
                                      const float* __restrict__ as,
                                      const float* __restrict__ ad,
                                      float* __restrict__ alpha,
                                      float* __restrict__ mbuf) {
  int idx = blockIdx.x * blockDim.x + threadIdx.x;
  if (idx >= N_EDGE * HEADS) return;
  int e = idx / HEADS, hh = idx % HEADS;
  float v = as[(size_t)src[e] * HEADS + hh] + ad[(size_t)dst[e] * HEADS + hh];
  v = v > 0.f ? v : 0.2f * v;                       // leaky_relu(0.2)
  alpha[idx] = v;
  atomicMaxF(&mbuf[(size_t)dst[e] * HEADS + hh], v);
}

__global__ void edge_exp_sum_kernel(const int* __restrict__ dst,
                                    float* __restrict__ alpha,   // -> exp
                                    const float* __restrict__ mbuf,
                                    float* __restrict__ sbuf) {
  int idx = blockIdx.x * blockDim.x + threadIdx.x;
  if (idx >= N_EDGE * HEADS) return;
  int e = idx / HEADS, hh = idx % HEADS;
  float ev = expf(alpha[idx] - mbuf[(size_t)dst[e] * HEADS + hh]);
  alpha[idx] = ev;
  atomicAdd(&sbuf[(size_t)dst[e] * HEADS + hh], ev);
}

// one block per edge, one lane per channel
__global__ void edge_scatter_kernel(const int* __restrict__ src,
                                    const int* __restrict__ dst,
                                    const float* __restrict__ hsrc,
                                    const float* __restrict__ eexp,
                                    const float* __restrict__ sbuf,
                                    float* __restrict__ o, int C) {
  int e = blockIdx.x;
  int c = threadIdx.x;
  if (c >= C) return;
  int D = C / HEADS;
  int hh = c / D;
  int sN = src[e], dN = dst[e];
  float a = eexp[(size_t)e * HEADS + hh] /
            (sbuf[(size_t)dN * HEADS + hh] + 1e-16f);
  atomicAdd(&o[(size_t)dN * C + c], hsrc[(size_t)sN * C + c] * a);
}

// --------------------------- semantic fusion -------------------------------

__global__ void semantic_score_kernel(const float* __restrict__ q,
                                      const float* __restrict__ colsum,
                                      float* __restrict__ wbuf, int N, int C) {
  __shared__ float red0[256], red1[256];
  int c = threadIdx.x;
  float v0 = 0.f, v1 = 0.f;
  if (c < C) { v0 = q[c] * colsum[c]; v1 = q[c] * colsum[C + c]; }
  red0[c] = v0; red1[c] = v1;
  __syncthreads();
  for (int st = 128; st >= 1; st >>= 1) {
    if (c < st) { red0[c] += red0[c + st]; red1[c] += red1[c + st]; }
    __syncthreads();
  }
  if (c == 0) {
    float s0 = red0[0] / (float)N, s1 = red1[0] / (float)N;
    float mx = fmaxf(s0, s1);
    float e0 = expf(s0 - mx), e1 = expf(s1 - mx);
    float inv = 1.f / (e0 + e1);
    wbuf[0] = e0 * inv; wbuf[1] = e1 * inv;
  }
}

__global__ void fuse2_kernel(const float* __restrict__ s0,
                             const float* __restrict__ s1,
                             const float* __restrict__ wbuf,
                             float* __restrict__ out, size_t n) {
  size_t i = (size_t)blockIdx.x * blockDim.x + threadIdx.x;
  if (i >= n) return;
  out[i] = wbuf[0] * s0[i] + wbuf[1] * s1[i];
}

// LayerNorm + ReLU, one wave32 per row
__global__ void __launch_bounds__(256)
ln_relu_kernel(float* __restrict__ x, const float* __restrict__ g,
               const float* __restrict__ b, int N, int C) {
  int lane = threadIdx.x & 31, wv = threadIdx.x >> 5;
  int row = blockIdx.x * 8 + wv;
  if (row >= N) return;
  float* p = x + (size_t)row * C;
  int per = C >> 5;                       // 8 (C=256) or 4 (C=128)
  float v[8];
  float sum = 0.f;
  for (int i = 0; i < per; ++i) { v[i] = p[lane + (i << 5)]; sum += v[i]; }
  for (int m = 16; m >= 1; m >>= 1) sum += __shfl_xor(sum, m, 32);
  float mu = sum / (float)C;
  float var = 0.f;
  for (int i = 0; i < per; ++i) { float d = v[i] - mu; var += d * d; }
  for (int m = 16; m >= 1; m >>= 1) var += __shfl_xor(var, m, 32);
  float rs = rsqrtf(var / (float)C + 1e-5f);
  for (int i = 0; i < per; ++i) {
    int c = lane + (i << 5);
    float y = (v[i] - mu) * rs * g[c] + b[c];
    p[c] = y > 0.f ? y : 0.f;
  }
}

__global__ void classifier_kernel(const float* __restrict__ x,
                                  const float* __restrict__ W,
                                  const float* __restrict__ bias,
                                  float* __restrict__ out, int N, int C) {
  int n = blockIdx.x * blockDim.x + threadIdx.x;
  if (n >= N) return;
  float a0 = bias[0], a1 = bias[1];
  const float* p = x + (size_t)n * C;
  for (int k = 0; k < C; ++k) { a0 += p[k] * W[k * 2]; a1 += p[k] * W[k * 2 + 1]; }
  out[(size_t)n * 2] = a0;
  out[(size_t)n * 2 + 1] = a1;
}

// --------------------------------- host side -------------------------------

static inline int cdiv(long long a, long long b) { return (int)((a + b - 1) / b); }

struct LayerBufs {
  float *hA, *hT, *o0, *o1, *o2, *o3, *rA, *rT;
  float *asb, *adb, *mb, *sb, *alpha, *colsum, *wb;
};

static void run_layer(const float* xA, const float* xT, int Cin, int Cout,
                      const _Float16* WAh, const float* bA,
                      const _Float16* WTh, const float* bT,
                      const float* att_src, const float* att_dst,   // (4,H,D)
                      const _Float16* kWh, const float* kb, const float* q,
                      const float* ln_g, const float* ln_b,
                      const int* const* esrc, const int* const* edst,
                      const LayerBufs& B, hipStream_t stream) {
  const int D = Cout / HEADS;
  // 1) node projections (WMMA)
  {
    dim3 gA(cdiv(N_ADDR, 128), Cout / 64);
    gemm_wmma_kernel<<<gA, 256, 0, stream>>>(xA, WAh, bA, B.hA, N_ADDR, Cout, Cin, 0);
    dim3 gT(cdiv(N_TX, 128), Cout / 64);
    gemm_wmma_kernel<<<gT, 256, 0, stream>>>(xT, WTh, bT, B.hT, N_TX, Cout, Cin, 0);
  }
  // 2) per-edge-type GAT aggregation
  const float* hs[4] = { B.hA, B.hT, B.hA, B.hT };   // a2t, t2a, a2a, t2t
  const float* hd[4] = { B.hT, B.hA, B.hA, B.hT };
  const int    Ns[4] = { N_ADDR, N_TX, N_ADDR, N_TX };
  const int    Nd[4] = { N_TX, N_ADDR, N_ADDR, N_TX };
  float*       oo[4] = { B.o0, B.o1, B.o2, B.o3 };
  for (int i = 0; i < 4; ++i) {
    node_att_kernel<<<cdiv((long long)Ns[i] * HEADS, 256), 256, 0, stream>>>(
        hs[i], att_src + (size_t)i * HEADS * D, B.asb, Ns[i], D);
    node_att_kernel<<<cdiv((long long)Nd[i] * HEADS, 256), 256, 0, stream>>>(
        hd[i], att_dst + (size_t)i * HEADS * D, B.adb, Nd[i], D);
    size_t ndh = (size_t)Nd[i] * HEADS;
    fill_kernel<<<cdiv(ndh, 256), 256, 0, stream>>>(B.mb, -3.0e38f, ndh);
    fill_kernel<<<cdiv(ndh, 256), 256, 0, stream>>>(B.sb, 0.f, ndh);
    size_t noc = (size_t)Nd[i] * Cout;
    fill_kernel<<<cdiv(noc, 256), 256, 0, stream>>>(oo[i], 0.f, noc);
    int egrid = cdiv((long long)N_EDGE * HEADS, 256);
    edge_alpha_max_kernel<<<egrid, 256, 0, stream>>>(esrc[i], edst[i], B.asb,
                                                     B.adb, B.alpha, B.mb);
    fixmax_kernel<<<cdiv(ndh, 256), 256, 0, stream>>>(B.mb, ndh);
    edge_exp_sum_kernel<<<egrid, 256, 0, stream>>>(edst[i], B.alpha, B.mb, B.sb);
    edge_scatter_kernel<<<N_EDGE, Cout, 0, stream>>>(esrc[i], edst[i], hs[i],
                                                     B.alpha, B.sb, oo[i], Cout);
    relu_inplace_kernel<<<cdiv(noc, 256), 256, 0, stream>>>(oo[i], noc);
  }
  // 3) semantic attention + fusion + LN + ReLU, per node type
  //    addr metapaths: [o1(t2a), o2(a2a)]   tx metapaths: [o0(a2t), o3(t2t)]
  const float* stkA[2] = { B.o1, B.o2 };
  const float* stkT[2] = { B.o0, B.o3 };
  const int    Nn[2]   = { N_ADDR, N_TX };
  const float* const* stks[2] = { stkA, stkT };
  float*       res[2] = { B.rA, B.rT };
  for (int nt = 0; nt < 2; ++nt) {
    fill_kernel<<<1, 256, 0, stream>>>(B.colsum, 0.f, (size_t)2 * Cout);
    for (int m = 0; m < 2; ++m) {
      dim3 g(cdiv(Nn[nt], 128), Cout / 64);
      gemm_tanh_colsum_kernel<<<g, 256, 0, stream>>>(
          stks[nt][m], kWh, kb, B.colsum + (size_t)m * Cout, Nn[nt], Cout, Cout);
    }
    semantic_score_kernel<<<1, 256, 0, stream>>>(q, B.colsum, B.wb, Nn[nt], Cout);
    size_t n = (size_t)Nn[nt] * Cout;
    fuse2_kernel<<<cdiv(n, 256), 256, 0, stream>>>(stks[nt][0], stks[nt][1],
                                                   B.wb, res[nt], n);
    ln_relu_kernel<<<cdiv(Nn[nt], 8), 256, 0, stream>>>(res[nt], ln_g, ln_b,
                                                        Nn[nt], Cout);
  }
}

extern "C" void kernel_launch(void* const* d_in, const int* in_sizes, int n_in,
                              void* d_out, int out_size, void* d_ws, size_t ws_size,
                              hipStream_t stream) {
  (void)in_sizes; (void)n_in; (void)out_size; (void)ws_size;
  const float* x_addr   = (const float*)d_in[0];
  const float* x_tx     = (const float*)d_in[1];
  const int*   esrc[4]  = { (const int*)d_in[2], (const int*)d_in[4],
                            (const int*)d_in[6], (const int*)d_in[8] };
  const int*   edst[4]  = { (const int*)d_in[3], (const int*)d_in[5],
                            (const int*)d_in[7], (const int*)d_in[9] };
  const float* W1_addr  = (const float*)d_in[10];
  const float* b1_addr  = (const float*)d_in[11];
  const float* W1_tx    = (const float*)d_in[12];
  const float* b1_tx    = (const float*)d_in[13];
  const float* att1_src = (const float*)d_in[14];
  const float* att1_dst = (const float*)d_in[15];
  const float* k1_W     = (const float*)d_in[16];
  const float* k1_b     = (const float*)d_in[17];
  const float* q1       = (const float*)d_in[18];
  const float* W2_addr  = (const float*)d_in[19];
  const float* b2_addr  = (const float*)d_in[20];
  const float* W2_tx    = (const float*)d_in[21];
  const float* b2_tx    = (const float*)d_in[22];
  const float* att2_src = (const float*)d_in[23];
  const float* att2_dst = (const float*)d_in[24];
  const float* k2_W     = (const float*)d_in[25];
  const float* k2_b     = (const float*)d_in[26];
  const float* q2       = (const float*)d_in[27];
  const float* ln1_g    = (const float*)d_in[28];
  const float* ln1_b    = (const float*)d_in[29];
  const float* ln2_g    = (const float*)d_in[30];
  const float* ln2_b    = (const float*)d_in[31];
  const float* lin_W    = (const float*)d_in[32];
  const float* lin_b    = (const float*)d_in[33];
  float* out = (float*)d_out;

  // ---- workspace carve (bump allocator, 256B aligned) ----
  char* ws = (char*)d_ws;
  size_t off = 0;
  auto alloc = [&](size_t bytes) -> void* {
    void* p = ws + off;
    off += (bytes + 255) & ~(size_t)255;
    return p;
  };
  _Float16* W1a_h = (_Float16*)alloc((size_t)F_IN * HID * 2);    // [N,K] = [HID,F_IN]
  _Float16* W1t_h = (_Float16*)alloc((size_t)F_IN * HID * 2);
  _Float16* W2a_h = (_Float16*)alloc((size_t)HID * OUT_C * 2);   // [OUT,HID]
  _Float16* W2t_h = (_Float16*)alloc((size_t)HID * OUT_C * 2);
  _Float16* k1W_h = (_Float16*)alloc((size_t)HID * HID * 2);
  _Float16* k2W_h = (_Float16*)alloc((size_t)OUT_C * OUT_C * 2);
  LayerBufs B;
  B.hA  = (float*)alloc((size_t)N_ADDR * HID * 4);    // reused for layer 2
  B.hT  = (float*)alloc((size_t)N_TX   * HID * 4);
  B.o0  = (float*)alloc((size_t)N_TX   * HID * 4);
  B.o1  = (float*)alloc((size_t)N_ADDR * HID * 4);
  B.o2  = (float*)alloc((size_t)N_ADDR * HID * 4);
  B.o3  = (float*)alloc((size_t)N_TX   * HID * 4);
  B.rA  = (float*)alloc((size_t)N_ADDR * HID * 4);
  B.rT  = (float*)alloc((size_t)N_TX   * HID * 4);
  B.asb = (float*)alloc((size_t)N_TX * HEADS * 4);
  B.adb = (float*)alloc((size_t)N_TX * HEADS * 4);
  B.mb  = (float*)alloc((size_t)N_TX * HEADS * 4);
  B.sb  = (float*)alloc((size_t)N_TX * HEADS * 4);
  B.alpha  = (float*)alloc((size_t)N_EDGE * HEADS * 4);
  B.colsum = (float*)alloc((size_t)2 * HID * 4);
  B.wb     = (float*)alloc((size_t)2 * 4);

  // ---- weight conversion to transposed f16 [N,K] (once per launch) ----
  f32_to_f16_T_kernel<<<cdiv(F_IN * HID, 256), 256, 0, stream>>>(W1_addr, W1a_h, F_IN, HID);
  f32_to_f16_T_kernel<<<cdiv(F_IN * HID, 256), 256, 0, stream>>>(W1_tx,   W1t_h, F_IN, HID);
  f32_to_f16_T_kernel<<<cdiv(HID * OUT_C, 256), 256, 0, stream>>>(W2_addr, W2a_h, HID, OUT_C);
  f32_to_f16_T_kernel<<<cdiv(HID * OUT_C, 256), 256, 0, stream>>>(W2_tx,   W2t_h, HID, OUT_C);
  f32_to_f16_T_kernel<<<cdiv(HID * HID, 256), 256, 0, stream>>>(k1_W, k1W_h, HID, HID);
  f32_to_f16_T_kernel<<<cdiv(OUT_C * OUT_C, 256), 256, 0, stream>>>(k2_W, k2W_h, OUT_C, OUT_C);

  // ---- layer 1 : F_IN -> HID ----
  run_layer(x_addr, x_tx, F_IN, HID, W1a_h, b1_addr, W1t_h, b1_tx,
            att1_src, att1_dst, k1W_h, k1_b, q1, ln1_g, ln1_b,
            esrc, edst, B, stream);

  // ---- layer 2 : HID -> OUT (rA/rT inputs are consumed by the projection
  //      GEMMs before fusion overwrites them — stream-ordered, safe reuse) ----
  run_layer(B.rA, B.rT, HID, OUT_C, W2a_h, b2_addr, W2t_h, b2_tx,
            att2_src, att2_dst, k2W_h, k2_b, q2, ln2_g, ln2_b,
            esrc, edst, B, stream);

  // ---- classifier on res2_addr ----
  classifier_kernel<<<cdiv(N_ADDR, 256), 256, 0, stream>>>(B.rA, lin_W, lin_b,
                                                           out, N_ADDR, OUT_C);
}
